// CrossAttention_77592879169834
// MI455X (gfx1250) — compile-verified, matrix operand order checked by reading, not run on hardware
//
#include <hip/hip_runtime.h>
#include <hip/hip_bf16.h>

// ---------------------------------------------------------------------------
// CDNA5 (gfx1250) gated cross-attention.
// All GEMMs via v_wmma_f32_16x16x32_bf16; bf16 tiles staged through LDS with
// async global->LDS copies where available; weights pre-transposed to [n][k].
// ---------------------------------------------------------------------------

typedef __attribute__((ext_vector_type(16))) __bf16 v16bf;
typedef __attribute__((ext_vector_type(8)))  float  v8f;
typedef __attribute__((ext_vector_type(8)))  unsigned short v8us;
typedef int v4i __attribute__((vector_size(16)));   // matches builtin param type

#define DEVINL __device__ __forceinline__

#if defined(__gfx1250__) && __has_builtin(__builtin_amdgcn_global_load_async_to_lds_b128)
#define ASYNC_LDS 1
#else
#define ASYNC_LDS 0
#endif

DEVINL unsigned short f2bf(float f) {                   // fp32 -> bf16 (RNE)
    unsigned u = __float_as_uint(f);
    return (unsigned short)((u + 0x7FFFu + ((u >> 16) & 1u)) >> 16);
}

// 16-byte global->LDS copy: async on gfx1250 when builtin exists, else sync.
DEVINL void copy16B(const unsigned short* g, unsigned short* l) {
#if ASYNC_LDS
    __builtin_amdgcn_global_load_async_to_lds_b128(
        (__attribute__((address_space(1))) v4i*)
            (__attribute__((address_space(1))) void*)const_cast<unsigned short*>(g),
        (__attribute__((address_space(3))) v4i*)
            (__attribute__((address_space(3))) void*)l,
        0, 0);
#else
    *(v8us*)l = *(const v8us*)g;
#endif
}

DEVINL void wait_async() {
#if ASYNC_LDS
#if __has_builtin(__builtin_amdgcn_s_wait_asynccnt)
    __builtin_amdgcn_s_wait_asynccnt(0);
#else
    asm volatile("s_wait_asynccnt 0" ::: "memory");
#endif
#endif
}

enum { EPI_BF16 = 0, EPI_BF16T = 1, EPI_F32SCALE = 2, EPI_BOTH = 3, EPI_GATE = 4 };

constexpr int BM = 128;     // block tile M
constexpr int BN = 128;     // block tile N
constexpr int KT = 32;      // K step (one bf16 WMMA deep)
constexpr int AS = KT + 8;  // padded LDS row stride (80B -> conflict-free b128)

// GEMM engine: C = [A|A2] @ B^T(+bias), B always bf16 stored [n][k] (ld=ldb).
//   TA  : float (converted to bf16 during LDS fill) or unsigned short (bf16)
//   EPI : epilogue selector
template <typename TA, int EPI>
__global__ __launch_bounds__(256)
void gemm_wmma(const TA* __restrict__ A, const TA* __restrict__ A2,
               const unsigned short* __restrict__ Bm, const float* __restrict__ bias,
               void* __restrict__ C0, void* __restrict__ C1,
               const float* __restrict__ xf, const float* __restrict__ maskv,
               const float* __restrict__ qres,
               int M, int N, int K, int K1,
               int lda, int ldb, int ldc, float scale,
               long long sA, long long sB, long long sC)
{
    __shared__ alignas(16) unsigned short AsT[BM * AS];
    __shared__ alignas(16) unsigned short BsT[BN * AS];

    const int t    = threadIdx.x;
    const int lane = t & 31;
    const int wave = t >> 5;
    const int mW   = (wave >> 1) * 32;   // wave tile: 32 (M) x 64 (N)
    const int nW   = (wave & 1) * 64;
    const int m0   = blockIdx.x * BM;
    const int n0   = blockIdx.y * BN;

    const TA* Ab             = A  + (size_t)blockIdx.z * sA;
    const unsigned short* Bb = Bm + (size_t)blockIdx.z * sB;

    v8f acc[2][4];
    #pragma unroll
    for (int i = 0; i < 2; ++i)
        #pragma unroll
        for (int j = 0; j < 4; ++j) { v8f z = {}; acc[i][j] = z; }

    const int fr = t >> 1;          // LDS fill: row handled by this thread
    const int fk = (t & 1) * 16;    // LDS fill: k-chunk base (16 elements)

    for (int k0 = 0; k0 < K; k0 += KT) {
        // ---- A tile fill: chunk-level concat select (K1 % 16 == 0) ----
        {
            const size_t rowG = (size_t)(m0 + fr);
            const int kgb = k0 + fk;
            const TA* srcp = (kgb < K1) ? (Ab + rowG * lda + kgb)
                                        : (A2 + rowG * lda + (kgb - K1));
            unsigned short* dst = &AsT[fr * AS + fk];
            if constexpr (sizeof(TA) == 2) {
                copy16B((const unsigned short*)srcp,     dst);
                copy16B((const unsigned short*)srcp + 8, dst + 8);
            } else {
                const float4* f4 = (const float4*)srcp;   // 16B aligned
                const float4 a0 = f4[0], a1 = f4[1], a2 = f4[2], a3 = f4[3];
                v8us lo, hi;
                lo[0]=f2bf(a0.x); lo[1]=f2bf(a0.y); lo[2]=f2bf(a0.z); lo[3]=f2bf(a0.w);
                lo[4]=f2bf(a1.x); lo[5]=f2bf(a1.y); lo[6]=f2bf(a1.z); lo[7]=f2bf(a1.w);
                hi[0]=f2bf(a2.x); hi[1]=f2bf(a2.y); hi[2]=f2bf(a2.z); hi[3]=f2bf(a2.w);
                hi[4]=f2bf(a3.x); hi[5]=f2bf(a3.y); hi[6]=f2bf(a3.z); hi[7]=f2bf(a3.w);
                *(v8us*)dst       = lo;
                *(v8us*)(dst + 8) = hi;
                if (k0 + KT < K)   // gfx1250 global_prefetch of next fp32 chunk
                    __builtin_prefetch((const void*)(srcp + KT), 0, 1);
            }
        }
        // ---- B tile fill: always bf16 [n][k], contiguous 16B chunks ----
        {
            const size_t nG = (size_t)(n0 + fr);
            const unsigned short* srcb = Bb + nG * ldb + (k0 + fk);
            unsigned short* dstb = &BsT[fr * AS + fk];
            copy16B(srcb,     dstb);
            copy16B(srcb + 8, dstb + 8);
        }
        wait_async();
        __syncthreads();

        // ---- fragment loads per ISA 16-bit A/B layout ----
        const int rsel = lane & 15;
        const int kb   = (lane >> 4) * 8;
        union Frag { v8us h[2]; v16bf v; };
        v16bf afr[2], bfr[4];
        #pragma unroll
        for (int mt = 0; mt < 2; ++mt) {
            const unsigned short* p = &AsT[(mW + mt * 16 + rsel) * AS + kb];
            Frag u; u.h[0] = *(const v8us*)p; u.h[1] = *(const v8us*)(p + 16);
            afr[mt] = u.v;
        }
        #pragma unroll
        for (int nt = 0; nt < 4; ++nt) {
            const unsigned short* p = &BsT[(nW + nt * 16 + rsel) * AS + kb];
            Frag u; u.h[0] = *(const v8us*)p; u.h[1] = *(const v8us*)(p + 16);
            bfr[nt] = u.v;
        }
        // ---- 8 WMMAs per wave per K-step ----
        #pragma unroll
        for (int mt = 0; mt < 2; ++mt)
            #pragma unroll
            for (int nt = 0; nt < 4; ++nt)
                acc[mt][nt] = __builtin_amdgcn_wmma_f32_16x16x32_bf16(
                    false, afr[mt], false, bfr[nt], (short)0, acc[mt][nt],
                    false, false);
        __syncthreads();
    }

    // ---- epilogue: C/D layout VGPR e -> M = e + (lane>=16)*8, N = lane&15 ----
    const size_t cbase = (size_t)blockIdx.z * sC;
    #pragma unroll
    for (int mt = 0; mt < 2; ++mt) {
        #pragma unroll
        for (int nt = 0; nt < 4; ++nt) {
            const int col  = n0 + nW + nt * 16 + (lane & 15);
            const int rowB = m0 + mW + mt * 16 + ((lane >> 4) * 8);
            #pragma unroll
            for (int e = 0; e < 8; ++e) {
                const int row = rowB + e;
                float v = acc[mt][nt][e];
                const size_t idx = cbase + (size_t)row * ldc + col;
                if constexpr (EPI == EPI_BF16) {
                    ((unsigned short*)C0)[idx] = f2bf(v + bias[col]);
                } else if constexpr (EPI == EPI_BF16T) {
                    // store transposed per batch: [b][d][l], b = row>>11, l = row&2047
                    const size_t tix = ((size_t)(row >> 11) * 1024 + col) * 2048
                                     + (size_t)(row & 2047);
                    ((unsigned short*)C0)[tix] = f2bf(v + bias[col]);
                } else if constexpr (EPI == EPI_F32SCALE) {
                    ((float*)C0)[idx] = v * scale;
                } else if constexpr (EPI == EPI_BOTH) {
                    ((float*)C0)[idx]          = v;
                    ((unsigned short*)C1)[idx] = f2bf(v);
                } else { // EPI_GATE: out = x * mask * sigmoid(g) + q
                    const float g  = v + bias[col];
                    const float sg = 1.0f / (1.0f + __expf(-g));
                    ((float*)C0)[idx] = xf[idx] * maskv[row] * sg + qres[idx];
                }
            }
        }
    }
}

// fp32 [K][N] -> bf16 [N][K] tiled transpose (32x32 tiles, 256 threads).
__global__ __launch_bounds__(256)
void transpose_to_bf16(const float* __restrict__ W, unsigned short* __restrict__ WT,
                       int K, int N)
{
    __shared__ float tile[32][33];
    const int tx  = threadIdx.x & 31;
    const int ty  = threadIdx.x >> 5;      // 0..7
    const int n0b = blockIdx.x * 32;
    const int k0b = blockIdx.y * 32;
    #pragma unroll
    for (int j = 0; j < 4; ++j)
        tile[ty + j * 8][tx] = W[(size_t)(k0b + ty + j * 8) * N + (n0b + tx)];
    __syncthreads();
    #pragma unroll
    for (int j = 0; j < 4; ++j)
        WT[(size_t)(n0b + ty + j * 8) * K + (k0b + tx)] = f2bf(tile[tx][ty + j * 8]);
}

// Row softmax over Lk=2048 elements, in place, one block per row.
__global__ __launch_bounds__(256)
void softmax_rows(float* __restrict__ S, int Lk)
{
    float* p = S + (size_t)blockIdx.x * Lk;
    const int t = threadIdx.x;
    float vloc[8];
    float vmax = -3.4e38f;
    #pragma unroll
    for (int i = 0; i < 8; ++i) {
        vloc[i] = p[t + i * 256];
        vmax = fmaxf(vmax, vloc[i]);
    }
    __shared__ float red[256];
    red[t] = vmax; __syncthreads();
    for (int s = 128; s > 0; s >>= 1) {
        if (t < s) red[t] = fmaxf(red[t], red[t + s]);
        __syncthreads();
    }
    vmax = red[0]; __syncthreads();
    float sum = 0.f;
    #pragma unroll
    for (int i = 0; i < 8; ++i) { vloc[i] = __expf(vloc[i] - vmax); sum += vloc[i]; }
    red[t] = sum; __syncthreads();
    for (int s = 128; s > 0; s >>= 1) {
        if (t < s) red[t] += red[t + s];
        __syncthreads();
    }
    const float inv = 1.0f / red[0];
    #pragma unroll
    for (int i = 0; i < 8; ++i) p[t + i * 256] = vloc[i] * inv;
}

extern "C" void kernel_launch(void* const* d_in, const int* in_sizes, int n_in,
                              void* d_out, int out_size, void* d_ws, size_t ws_size,
                              hipStream_t stream)
{
    const float* q    = (const float*)d_in[0];
    const float* k    = (const float*)d_in[1];
    const float* v    = (const float*)d_in[2];
    const float* mask = (const float*)d_in[3];
    const float* Wq   = (const float*)d_in[4];
    const float* bq   = (const float*)d_in[5];
    const float* Wk   = (const float*)d_in[6];
    const float* bk   = (const float*)d_in[7];
    const float* Wv   = (const float*)d_in[8];
    const float* bv   = (const float*)d_in[9];
    const float* Wg   = (const float*)d_in[10];
    const float* bg   = (const float*)d_in[11];
    float* out = (float*)d_out;

    constexpr int Bn = 8, L = 2048, D = 1024;
    constexpr int M  = Bn * L;                 // 16384 rows
    const size_t nProj = (size_t)M * D;        // 16M elements

    // workspace layout (~330 MB)
    char* ws = (char*)d_ws;
    unsigned short* qp  = (unsigned short*)ws;  ws += nProj * 2;        // Qp  bf16
    unsigned short* kp  = (unsigned short*)ws;  ws += nProj * 2;        // Kp  bf16
    unsigned short* vpT = (unsigned short*)ws;  ws += nProj * 2;        // Vp^T bf16 [b][d][l]
    unsigned short* xb  = (unsigned short*)ws;  ws += nProj * 2;        // X   bf16
    float* xf = (float*)ws;                     ws += nProj * 4;        // X   fp32
    float* S  = (float*)ws;                     ws += (size_t)Bn*L*L*4; // scores fp32
    unsigned short* WqT = (unsigned short*)ws;  ws += (size_t)D * D * 2;
    unsigned short* WkT = (unsigned short*)ws;  ws += (size_t)D * D * 2;
    unsigned short* WvT = (unsigned short*)ws;  ws += (size_t)D * D * 2;
    unsigned short* WgT = (unsigned short*)ws;  // 2D x D

    const dim3 blk(256);

    // 0) weights -> bf16, transposed to [n][k]
    transpose_to_bf16<<<dim3(D/32, D/32),     blk, 0, stream>>>(Wq, WqT, D,     D);
    transpose_to_bf16<<<dim3(D/32, D/32),     blk, 0, stream>>>(Wk, WkT, D,     D);
    transpose_to_bf16<<<dim3(D/32, D/32),     blk, 0, stream>>>(Wv, WvT, D,     D);
    transpose_to_bf16<<<dim3(D/32, (2*D)/32), blk, 0, stream>>>(Wg, WgT, 2 * D, D);

    // 1) projections: qp/kp = proj(q/k); vp stored transposed [b][d][l]
    const dim3 gProj(M / BM, D / BN, 1);
    gemm_wmma<float, EPI_BF16><<<gProj, blk, 0, stream>>>(
        q, q, WqT, bq, qp, nullptr, nullptr, nullptr, nullptr,
        M, D, D, D, D, D, D, 1.0f, 0, 0, 0);
    gemm_wmma<float, EPI_BF16><<<gProj, blk, 0, stream>>>(
        k, k, WkT, bk, kp, nullptr, nullptr, nullptr, nullptr,
        M, D, D, D, D, D, D, 1.0f, 0, 0, 0);
    gemm_wmma<float, EPI_BF16T><<<gProj, blk, 0, stream>>>(
        v, v, WvT, bv, vpT, nullptr, nullptr, nullptr, nullptr,
        M, D, D, D, D, D, D, 1.0f, 0, 0, 0);

    // 2) scores: S = (1/sqrt(D)) * Qp @ Kp^T per batch (Kp already [n][k])
    const dim3 gS(L / BM, L / BN, Bn);
    gemm_wmma<unsigned short, EPI_F32SCALE><<<gS, blk, 0, stream>>>(
        qp, qp, kp, nullptr, S, nullptr, nullptr, nullptr, nullptr,
        L, L, D, D, D, D, L, 0.03125f,
        (long long)L * D, (long long)L * D, (long long)L * L);

    // 3) softmax rows (in place, fp32)
    softmax_rows<<<dim3(M), blk, 0, stream>>>(S, L);

    // 4) X = P @ Vp per batch (Vp^T is [d][l] => B_NT layout, ldb = L)
    const dim3 gX(L / BM, D / BN, Bn);
    gemm_wmma<float, EPI_BOTH><<<gX, blk, 0, stream>>>(
        S, S, vpT, nullptr, xf, xb, nullptr, nullptr, nullptr,
        L, D, L, L, L, L, D, 1.0f,
        (long long)L * L, (long long)D * L, (long long)L * D);

    // 5) gate = [qp | x] @ Wg + bg ; out = x * mask * sigmoid(gate) + q
    gemm_wmma<unsigned short, EPI_GATE><<<gProj, blk, 0, stream>>>(
        qp, xb, WgT, bg, out, nullptr, xf, mask, q,
        M, D, 2 * D, D, D, 2 * D, D, 1.0f, 0, 0, 0);
}
